// SetConvNoDown_46299747450893
// MI455X (gfx1250) — matrix operand order repaired and checked by reading, not compiled, not run on hardware
//
#include <hip/hip_runtime.h>

typedef __attribute__((ext_vector_type(16))) _Float16 v16h;
typedef __attribute__((ext_vector_type(8)))  _Float16 v8h;
typedef __attribute__((ext_vector_type(8)))  float    v8f;
typedef __attribute__((ext_vector_type(4)))  unsigned int u32x4;
typedef __attribute__((ext_vector_type(8)))  int i32x8;
typedef __attribute__((ext_vector_type(4)))  int i32x4;

#define Bsz    8
#define Npts   2048
#define INCH   64
#define NPOINT (Bsz * Npts)   // 16384

// ---------------------------------------------------------------- helpers ---

__device__ __forceinline__ v8f zero8() {
  v8f z;
#pragma unroll
  for (int r = 0; r < 8; ++r) z[r] = 0.f;
  return z;
}

// A-matrix fragment (16x32 f16): lane m = lane&15, half = lane>>4;
// element s holds K = half*16 + s (contiguous, matches B fragment).
__device__ __forceinline__ v16h load_a_frag(const _Float16* base, int stride,
                                            int kbase, int lane) {
  int m = lane & 15, h = (lane >> 4) & 1;
  const _Float16* row = base + m * stride + kbase + h * 16;
  v8h lo = *(const v8h*)(row);
  v8h hi = *(const v8h*)(row + 8);
  v16h r;
#pragma unroll
  for (int e = 0; e < 8; ++e) { r[e] = lo[e]; r[e + 8] = hi[e]; }
  return r;
}

// B-matrix fragment (32x16 f16): lane n = lane&15 is the output column
// (= weight row), K = kbase + half*16 + 0..15 contiguous -> one 32B load.
__device__ __forceinline__ v16h load_b_frag(const _Float16* w, int stride,
                                            int nt, int kbase, int lane) {
  int n = lane & 15, h = (lane >> 4) & 1;
  return *(const v16h*)(w + (size_t)(nt * 16 + n) * stride + kbase + h * 16);
}

__device__ __forceinline__ v8f wmma_f16(v16h a, v16h b, v8f c) {
  return __builtin_amdgcn_wmma_f32_16x16x32_f16(false, a, false, b,
                                                (short)0, c, false, false);
}

// Async global->LDS copy, 16B per lane (ASYNCcnt path).
__device__ __forceinline__ void async_copy_b128(unsigned lds_byte_addr,
                                                const void* gaddr) {
  asm volatile("global_load_async_to_lds_b128 %0, %1, off"
               :: "v"(lds_byte_addr),
                  "v"((unsigned long long)(uintptr_t)gaddr)
               : "memory");
}
__device__ __forceinline__ void wait_asynccnt0() {
  asm volatile("s_wait_asynccnt 0" ::: "memory");
}

// Generic LDS pointer -> wave-relative LDS byte address (flat LDS aperture
// keeps the LDS offset in addr[31:0]).
__device__ __forceinline__ unsigned lds_addr32(const void* p) {
  return (unsigned)(uintptr_t)p;
}

// ------------------------------------------------------------- prep kernels -

__global__ void fold_bn_kernel(const float* __restrict__ gamma,
                               const float* __restrict__ beta,
                               const float* __restrict__ mean,
                               const float* __restrict__ var,
                               float* __restrict__ sb, int C) {
  int c = blockIdx.x * blockDim.x + threadIdx.x;
  if (c < C) {
    float s = gamma[c] * rsqrtf(var[c] + 1e-5f);
    sb[c] = s;
    sb[C + c] = beta[c] - mean[c] * s;
  }
}

__global__ void convert_f16_kernel(const float* __restrict__ src,
                                   _Float16* __restrict__ out, int nelem) {
  int i = blockIdx.x * blockDim.x + threadIdx.x;
  if (i < nelem) out[i] = (_Float16)src[i];
}

// W is (64 x 67) row-major, input = [rel_xyz(3), feat(64)].
// Repack to (64 x 96): cols 0..63 = feat weights, 64..66 = xyz weights,
// 67..95 = 0 (so the A-tile padding never contributes).
__global__ void convert_w1_kernel(const float* __restrict__ W,
                                  _Float16* __restrict__ out) {
  int i = blockIdx.x * blockDim.x + threadIdx.x;
  if (i < 64 * 96) {
    int r = i / 96, c = i % 96;
    float v = 0.f;
    if (c < 64)      v = W[r * 67 + 3 + c];
    else if (c < 67) v = W[r * 67 + (c - 64)];
    out[i] = (_Float16)v;
  }
}

// -------------------------------------------------------------- ball query -

// score = j if d2 < r^2 else n; top_k(-score) == first k in-radius indices,
// padded with the first hit (self is always a hit: d2(i,i)=0).
__global__ void ballquery_kernel(const float* __restrict__ xyz,
                                 int* __restrict__ idx16,
                                 int* __restrict__ idx32) {
  __shared__ float sx[256], sy[256], sz[256];
  int b = blockIdx.x >> 3;                       // 8 blocks per batch
  int i = ((blockIdx.x & 7) << 8) + threadIdx.x;
  const float* xb = xyz + (size_t)b * Npts * 3;
  float px = xb[i * 3 + 0], py = xb[i * 3 + 1], pz = xb[i * 3 + 2];
  int c16 = 0, c32 = 0, f16i = -1, f32i = -1;
  int* o16 = idx16 + (size_t)(b * Npts + i) * 16;
  int* o32 = idx32 + (size_t)(b * Npts + i) * 32;
  for (int t = 0; t < Npts; t += 256) {
    __syncthreads();
    int j = t + threadIdx.x;
    sx[threadIdx.x] = xb[j * 3 + 0];
    sy[threadIdx.x] = xb[j * 3 + 1];
    sz[threadIdx.x] = xb[j * 3 + 2];
    __syncthreads();
    if (c16 < 16 || c32 < 32) {
      for (int u = 0; u < 256; ++u) {
        float dx = sx[u] - px, dy = sy[u] - py, dz = sz[u] - pz;
        float d2 = dx * dx + dy * dy + dz * dz;
        if (d2 < 0.04f) {
          int jj = t + u;
          if (c32 < 32) { if (f32i < 0) f32i = jj; o32[c32++] = jj; }
          if (d2 < 0.01f && c16 < 16) { if (f16i < 0) f16i = jj; o16[c16++] = jj; }
        }
      }
    }
  }
  for (int s = c16; s < 16; ++s) o16[s] = f16i;
  for (int s = c32; s < 32; ++s) o32[s] = f32i;
}

// ----------------------------------------------------------- branch kernel -

// One wave32 per point: feat rows pulled into the LDS A-tile with
// global_load_async_to_lds_b128 (4 rows x 128B per issue), rel_xyz + pad via
// DS, two WMMA layers with folded BN+ReLU, max over neighbors.
template <int KNB>
__global__ void __launch_bounds__(128) branch_kernel(
    const float* __restrict__ xyz,
    const _Float16* __restrict__ feat16,   // (B*N, 64) f16
    const int* __restrict__ idx,
    const _Float16* __restrict__ w1,       // 64 x 96 (padded, reordered)
    const _Float16* __restrict__ w2,       // 128 x 64
    const float* __restrict__ sb1,         // scale[64] | bias[64]
    const float* __restrict__ sb2,         // scale[128] | bias[128]
    _Float16* __restrict__ concat_out, int branch_off) {
  constexpr int MT = KNB / 16;
  __shared__ __align__(32) _Float16 shA[4][KNB * 96];
  __shared__ __align__(32) _Float16 shY[4][KNB * 64];
  int wv = threadIdx.x >> 5, lane = threadIdx.x & 31;
  int p = blockIdx.x * 4 + wv;                // point id, exact cover
  int b = p >> 11, i = p & (Npts - 1);
  _Float16* A = shA[wv];
  _Float16* Y = shY[wv];
  const float* xb = xyz + (size_t)b * Npts * 3;
  const _Float16* fb = feat16 + (size_t)b * Npts * INCH;
  const int* ip = idx + (size_t)p * KNB;

  __builtin_prefetch(w1, 0, 0);
  __builtin_prefetch(w2, 0, 0);

  // ---- gather feat rows (cols 0..63) via async LDS copies:
  // lane -> (row = lane>>3, 16B chunk = lane&7); 4 rows per issue.
  unsigned abase = lds_addr32(A);
  int grow = lane >> 3, gch = lane & 7;
  for (int s0 = 0; s0 < KNB; s0 += 4) {
    int j = ip[s0 + grow];
    const _Float16* src = fb + (size_t)j * INCH + gch * 8;
    unsigned dst = abase + (unsigned)(((s0 + grow) * 96 + gch * 8) * 2);
    async_copy_b128(dst, src);
  }
  // ---- rel_xyz (cols 64..66) + zero pad (cols 67..95) via DS
  for (int s = 0; s < KNB; ++s) {
    int j = ip[s];
    _Float16 t = (_Float16)0.f;
    if (lane < 3) t = (_Float16)(xb[j * 3 + lane] - xb[i * 3 + lane]);
    A[s * 96 + 64 + lane] = t;                // lanes 3..31 zero cols 67..95
  }
  wait_asynccnt0();                            // feat rows resident in LDS

  int n = lane & 15, h = (lane >> 4) & 1;

  // ---- layer 1: (KNB x 96) @ (96 -> 64), BN+ReLU, to LDS
  for (int nt = 0; nt < 4; ++nt) {
    v8f acc[MT];
#pragma unroll
    for (int mt = 0; mt < MT; ++mt) acc[mt] = zero8();
#pragma unroll
    for (int ks = 0; ks < 3; ++ks) {
      v16h bf = load_b_frag(w1, 96, nt, ks * 32, lane);
#pragma unroll
      for (int mt = 0; mt < MT; ++mt) {
        v16h af = load_a_frag(A + mt * 16 * 96, 96, ks * 32, lane);
        acc[mt] = wmma_f16(af, bf, acc[mt]);
      }
    }
    int co = nt * 16 + n;
    float sc = sb1[co], bi = sb1[64 + co];
#pragma unroll
    for (int mt = 0; mt < MT; ++mt)
#pragma unroll
      for (int r = 0; r < 8; ++r) {
        float v = fmaxf(acc[mt][r] * sc + bi, 0.f);
        Y[(mt * 16 + r + 8 * h) * 64 + co] = (_Float16)v;
      }
  }

  // ---- layer 2: (KNB x 64) @ (64 -> 128), BN+ReLU, max over neighbors
  for (int nt = 0; nt < 8; ++nt) {
    v8f acc[MT];
#pragma unroll
    for (int mt = 0; mt < MT; ++mt) acc[mt] = zero8();
#pragma unroll
    for (int ks = 0; ks < 2; ++ks) {
      v16h bf = load_b_frag(w2, 64, nt, ks * 32, lane);
#pragma unroll
      for (int mt = 0; mt < MT; ++mt) {
        v16h af = load_a_frag(Y + mt * 16 * 64, 64, ks * 32, lane);
        acc[mt] = wmma_f16(af, bf, acc[mt]);
      }
    }
    int co = nt * 16 + n;
    float sc = sb2[co], bi = sb2[128 + co];
    float mx = 0.f;                           // ReLU output >= 0
#pragma unroll
    for (int mt = 0; mt < MT; ++mt)
#pragma unroll
      for (int r = 0; r < 8; ++r)
        mx = fmaxf(mx, fmaxf(acc[mt][r] * sc + bi, 0.f));
    mx = fmaxf(mx, __shfl_xor(mx, 16, 32));   // merge the two lane halves
    if (lane < 16)
      concat_out[(size_t)p * 256 + branch_off + co] = (_Float16)mx;
  }
}

// ------------------------------------------------------------- fuse kernel -

__global__ void __launch_bounds__(128) fuse_kernel(
    const _Float16* __restrict__ xin,  // 16384 x 256
    const _Float16* __restrict__ w3,   // 128 x 256 (global, L2 resident)
    const _Float16* __restrict__ w4,   // 128 x 128 (TDM-staged into LDS)
    const float* __restrict__ sb3, const float* __restrict__ sb4,
    float* __restrict__ out) {         // 16384 x 128
  __shared__ __align__(32) _Float16 shY[4][16 * 128];
  __shared__ __align__(32) _Float16 shW[128 * 128];
  int wv = threadIdx.x >> 5, lane = threadIdx.x & 31;
  int t = blockIdx.x * 4 + wv;                 // 16-row tile id
  _Float16* Y = shY[wv];
  const _Float16* Ain = xin + (size_t)t * 16 * 256;
  int n = lane & 15, h = (lane >> 4) & 1;

  __builtin_prefetch(w3, 0, 0);

#if __has_builtin(__builtin_amdgcn_tensor_load_to_lds)
  // Stage w4 (32KB) into LDS with one Tensor-Data-Mover descriptor:
  // 1 row x 16384 elem, data_size=2B (D# layout per ISA 08_async_tensor 8.3-8.6).
  if (wv == 0) {
    unsigned long long ga = (unsigned long long)(uintptr_t)w4;
    u32x4 g0;
    g0[0] = 1u;                                   // count=1, no gather/restore
    g0[1] = lds_addr32(shW);                      // lds_addr
    g0[2] = (unsigned)(ga & 0xffffffffull);       // global_addr[31:0]
    g0[3] = (unsigned)((ga >> 32) & 0x1ffffffull) // global_addr[56:32]
            | (2u << 30);                         // type = 2 ("image")
    i32x8 g1;
    g1[0] = 0x10000;              // workgroup_mask=0, data_size=1 (2 bytes)
    g1[1] = (int)(16384u << 16);  // tensor_dim0 = 16384 (bits 79:48 low half)
    g1[2] = 0x10000;              // tensor_dim1 = 1 (bits 111:80 low half)
    g1[3] = (int)(16384u << 16);  // tile_dim0 = 16384 (bits 127:112)
    g1[4] = 1;                    // tile_dim1 = 1, tile_dim2 = 0
    g1[5] = 16384;                // tensor_dim0_stride[31:0]
    g1[6] = 0;                    // stride[47:32] | tensor_dim1_stride lo
    g1[7] = 0;
    i32x4 gz;
    gz[0] = 0; gz[1] = 0; gz[2] = 0; gz[3] = 0;
#if __has_include(<hip/amd_detail/amd_gfx1250_TDM.h>)
    i32x8 gz8;
#pragma unroll
    for (int e = 0; e < 8; ++e) gz8[e] = 0;
    __builtin_amdgcn_tensor_load_to_lds(g0, g1, gz, gz, gz8, 0);
#else
    __builtin_amdgcn_tensor_load_to_lds(g0, g1, gz, gz, 0);
#endif
    __builtin_amdgcn_s_wait_tensorcnt(0);
  }
#else
  for (int e = threadIdx.x; e < 128 * 128; e += 128) shW[e] = w4[e];
#endif
  __syncthreads();                               // w4 tile visible to all waves

  for (int nt = 0; nt < 8; ++nt) {               // 256 -> 128
    v8f acc = zero8();
#pragma unroll
    for (int ks = 0; ks < 8; ++ks) {
      v16h af = load_a_frag(Ain, 256, ks * 32, lane);
      v16h bf = load_b_frag(w3, 256, nt, ks * 32, lane);
      acc = wmma_f16(af, bf, acc);
    }
    int co = nt * 16 + n;
    float sc = sb3[co], bi = sb3[128 + co];
#pragma unroll
    for (int r = 0; r < 8; ++r) {
      float v = fmaxf(acc[r] * sc + bi, 0.f);
      Y[(r + 8 * h) * 128 + co] = (_Float16)v;
    }
  }
  for (int nt = 0; nt < 8; ++nt) {               // 128 -> 128 (B from LDS)
    v8f acc = zero8();
#pragma unroll
    for (int ks = 0; ks < 4; ++ks) {
      v16h af = load_a_frag(Y, 128, ks * 32, lane);
      v16h bf = load_b_frag(shW, 128, nt, ks * 32, lane);
      acc = wmma_f16(af, bf, acc);
    }
    int co = nt * 16 + n;
    float sc = sb4[co], bi = sb4[128 + co];
#pragma unroll
    for (int r = 0; r < 8; ++r)
      out[(size_t)(t * 16 + r + 8 * h) * 128 + co] =
          fmaxf(acc[r] * sc + bi, 0.f);
  }
}

// ------------------------------------------------------------------ launch -

extern "C" void kernel_launch(void* const* d_in, const int* in_sizes, int n_in,
                              void* d_out, int out_size, void* d_ws,
                              size_t ws_size, hipStream_t stream) {
  (void)in_sizes; (void)n_in; (void)out_size; (void)ws_size;
  const float* xyz  = (const float*)d_in[0];
  const float* feat = (const float*)d_in[1];
  // param leaves: per layer [W, beta, gamma, mean, var] (alphabetical);
  // layers in pytree order: br0.l0, br0.l1, br1.l0, br1.l1, fuse0, fuse1
  const float *W[6], *BE[6], *GA[6], *ME[6], *VA[6];
  for (int l = 0; l < 6; ++l) {
    W[l]  = (const float*)d_in[2 + 5 * l + 0];
    BE[l] = (const float*)d_in[2 + 5 * l + 1];
    GA[l] = (const float*)d_in[2 + 5 * l + 2];
    ME[l] = (const float*)d_in[2 + 5 * l + 3];
    VA[l] = (const float*)d_in[2 + 5 * l + 4];
  }

  char* ws = (char*)d_ws;
  size_t off = 0;
  auto take = [&](size_t bytes) -> void* {
    off = (off + 255) & ~(size_t)255;
    void* q = ws + off;
    off += bytes;
    return q;
  };
  int*      idx16  = (int*)take((size_t)NPOINT * 16 * 4);
  int*      idx32  = (int*)take((size_t)NPOINT * 32 * 4);
  _Float16* feat16 = (_Float16*)take((size_t)NPOINT * INCH * 2);
  _Float16* concat = (_Float16*)take((size_t)NPOINT * 256 * 2);
  _Float16* w1h    = (_Float16*)take((size_t)2 * 64 * 96 * 2);
  _Float16* w2h    = (_Float16*)take((size_t)2 * 128 * 64 * 2);
  _Float16* w3h    = (_Float16*)take((size_t)128 * 256 * 2);
  _Float16* w4h    = (_Float16*)take((size_t)128 * 128 * 2);
  float*    sb1    = (float*)take((size_t)2 * 128 * 4);
  float*    sb2    = (float*)take((size_t)2 * 256 * 4);
  float*    sb3    = (float*)take((size_t)256 * 4);
  float*    sb4    = (float*)take((size_t)256 * 4);

  // BN folding (scale|bias per cout)
  fold_bn_kernel<<<1, 64, 0, stream>>>(GA[0], BE[0], ME[0], VA[0], sb1, 64);
  fold_bn_kernel<<<1, 64, 0, stream>>>(GA[2], BE[2], ME[2], VA[2], sb1 + 128, 64);
  fold_bn_kernel<<<1, 128, 0, stream>>>(GA[1], BE[1], ME[1], VA[1], sb2, 128);
  fold_bn_kernel<<<1, 128, 0, stream>>>(GA[3], BE[3], ME[3], VA[3], sb2 + 256, 128);
  fold_bn_kernel<<<1, 128, 0, stream>>>(GA[4], BE[4], ME[4], VA[4], sb3, 128);
  fold_bn_kernel<<<1, 128, 0, stream>>>(GA[5], BE[5], ME[5], VA[5], sb4, 128);
  // f16 conversions: activations + weights (cin reorder/pad for layer-1 W)
  convert_f16_kernel<<<(NPOINT * INCH + 255) / 256, 256, 0, stream>>>(
      feat, feat16, NPOINT * INCH);
  convert_w1_kernel<<<(64 * 96 + 255) / 256, 256, 0, stream>>>(W[0], w1h);
  convert_w1_kernel<<<(64 * 96 + 255) / 256, 256, 0, stream>>>(W[2], w1h + 64 * 96);
  convert_f16_kernel<<<(128 * 64 + 255) / 256, 256, 0, stream>>>(W[1], w2h, 128 * 64);
  convert_f16_kernel<<<(128 * 64 + 255) / 256, 256, 0, stream>>>(W[3], w2h + 128 * 64, 128 * 64);
  convert_f16_kernel<<<(128 * 256 + 255) / 256, 256, 0, stream>>>(W[4], w3h, 128 * 256);
  convert_f16_kernel<<<(128 * 128 + 255) / 256, 256, 0, stream>>>(W[5], w4h, 128 * 128);
  // ball query (both radii in one pass)
  ballquery_kernel<<<Bsz * (Npts / 256), 256, 0, stream>>>(xyz, idx16, idx32);
  // branches (one wave per point)
  branch_kernel<16><<<NPOINT / 4, 128, 0, stream>>>(
      xyz, feat16, idx16, w1h, w2h, sb1, sb2, concat, 0);
  branch_kernel<32><<<NPOINT / 4, 128, 0, stream>>>(
      xyz, feat16, idx32, w1h + 64 * 96, w2h + 128 * 64, sb1 + 128, sb2 + 256,
      concat, 128);
  // fuse MLP (one wave per 16-row tile)
  fuse_kernel<<<NPOINT / 16 / 4, 128, 0, stream>>>(concat, w3h, w4h, sb3, sb4,
                                                   (float*)d_out);
}
